// MultiHeadSigmoidSelfAttention_62612033241693
// MI455X (gfx1250) — compile-verified, matrix operand order checked by reading, not run on hardware
//
#include <hip/hip_runtime.h>

typedef __attribute__((ext_vector_type(16))) _Float16 v16h;
typedef __attribute__((ext_vector_type(8)))  _Float16 v8h;
typedef __attribute__((ext_vector_type(8)))  float    v8f;

#define FEATS 768
#define NHEAD 12
#define HD    64
#define BATCH 8
#define SEQ   1024
#define M_TOT (BATCH * SEQ)   // 8192 rows for the projection GEMMs

union Frag16 { v16h v; v8h h[2]; };

// Load a 16-element f16 WMMA A/B fragment for one lane.
// ISA 16-bit A/B layout (wave32): lane group g = lane>>4 selects K chunks
//   elements 0..7  <- K = g*8 + 0..7
//   elements 8..15 <- K = 16 + g*8 + 0..7
// p must already point at (row_base + g*8); chunks are 16B-aligned.
__device__ __forceinline__ v16h load_frag(const _Float16* p) {
    Frag16 f;
    f.h[0] = *(const v8h*)(p);
    f.h[1] = *(const v8h*)(p + 16);
    return f.v;
}

__device__ __forceinline__ v8f wmma16(v16h a, v16h b, v8f c) {
    return __builtin_amdgcn_wmma_f32_16x16x32_f16(
        /*neg_a=*/false, a, /*neg_b=*/false, b,
        /*c_mod=*/(short)0, c, /*reuse_a=*/false, /*reuse_b=*/false);
}

// ---------------------------------------------------------------------------
// fp32 -> fp16 conversion
// ---------------------------------------------------------------------------
__global__ void cvt_f32_f16(const float* __restrict__ in,
                            _Float16* __restrict__ out, int n) {
    int i = blockIdx.x * 256 + threadIdx.x;
    if (i < n) out[i] = (_Float16)in[i];
}

// ---------------------------------------------------------------------------
// C(M x 768) = A(M x 768, f16) * W^T (W is 768x768 row-major, f16) + bias
// Per wave: 64 (rows) x 64 (cols) tile -> 16 WMMA per 32-wide K-step with
// only 8 fragment loads (0.5 frag-loads/WMMA).  Block = 4 waves arranged
// 2x2 covering a 128x128 output tile.
// mode 0/1: write f16 to (b,h,n,hd)   [Q, K]
// mode 2  : write f16 to (b,h,hd,n)   [V transposed]
// mode 3  : write f32 row-major to d_out, final projection
// ---------------------------------------------------------------------------
__global__ void __launch_bounds__(128)
gemm_xwT(const _Float16* __restrict__ A, const _Float16* __restrict__ W,
         const float* __restrict__ bias, _Float16* __restrict__ outh,
         float* __restrict__ outf, int mode) {
    const int lane = threadIdx.x & 31;
    const int wave = threadIdx.x >> 5;
    const int l16  = lane & 15;
    const int g    = lane >> 4;
    const int mbase = blockIdx.x * 128 + (wave & 1) * 64;
    const int nbase = blockIdx.y * 128 + (wave >> 1) * 64;

    v8f acc[4][4];
#pragma unroll
    for (int i = 0; i < 4; ++i)
#pragma unroll
        for (int j = 0; j < 4; ++j) acc[i][j] = {};

    const _Float16* ap[4];
    const _Float16* wp[4];
#pragma unroll
    for (int rg = 0; rg < 4; ++rg)
        ap[rg] = A + (size_t)(mbase + rg * 16 + l16) * FEATS + g * 8;
#pragma unroll
    for (int cg = 0; cg < 4; ++cg)
        wp[cg] = W + (size_t)(nbase + cg * 16 + l16) * FEATS + g * 8;

    for (int k0 = 0; k0 < FEATS; k0 += 32) {
        __builtin_prefetch(ap[0] + k0 + 128, 0, 1);   // global_prefetch_b8
        v16h a0 = load_frag(ap[0] + k0);
        v16h a1 = load_frag(ap[1] + k0);
        v16h a2 = load_frag(ap[2] + k0);
        v16h a3 = load_frag(ap[3] + k0);
#pragma unroll
        for (int cg = 0; cg < 4; ++cg) {
            v16h bf = load_frag(wp[cg] + k0);
            acc[0][cg] = wmma16(a0, bf, acc[0][cg]);
            acc[1][cg] = wmma16(a1, bf, acc[1][cg]);
            acc[2][cg] = wmma16(a2, bf, acc[2][cg]);
            acc[3][cg] = wmma16(a3, bf, acc[3][cg]);
        }
    }

    // C fragment layout: element r of v8f -> row m = mbase + rg*16 + g*8 + r,
    // col = nbase + cg*16 + l16
#pragma unroll
    for (int rg = 0; rg < 4; ++rg) {
#pragma unroll
        for (int cg = 0; cg < 4; ++cg) {
            const int col = nbase + cg * 16 + l16;
            const float bv = bias[col];
            const int h  = col >> 6;    // col / 64
            const int hd = col & 63;
#pragma unroll
            for (int r = 0; r < 8; ++r) {
                const int m  = mbase + rg * 16 + g * 8 + r;
                const float val = acc[rg][cg][r] + bv;
                if (mode == 3) {
                    outf[(size_t)m * FEATS + col] = val;
                } else {
                    const int bb = m >> 10;      // m / SEQ
                    const int ns = m & 1023;     // m % SEQ
                    size_t idx;
                    if (mode == 2)  // V transposed: (b,h,hd,n)
                        idx = (((size_t)(bb * NHEAD + h)) * HD + hd) * SEQ + ns;
                    else            // Q,K: (b,h,n,hd)
                        idx = (((size_t)(bb * NHEAD + h)) * SEQ + ns) * HD + hd;
                    outh[idx] = (_Float16)val;
                }
            }
        }
    }
}

// ---------------------------------------------------------------------------
// Fused sigmoid attention per (b,h): each wave owns 32 Q rows, streams K/V in
// 32-column tiles.  S = sigmoid(Q K^T / sqrt(768) + bias) computed via WMMA,
// staged through LDS as f16 A-fragments, then O += S * V via WMMA (V is
// pre-transposed so its B-fragments are contiguous loads).
// ---------------------------------------------------------------------------
#define SROW 40   // LDS row stride in halves (80B, keeps 16B chunk alignment)

__global__ void __launch_bounds__(128)
sig_attn(const _Float16* __restrict__ Qh, const _Float16* __restrict__ Kh,
         const _Float16* __restrict__ Vt, const float* __restrict__ biasp,
         _Float16* __restrict__ Ao) {
    const int lane = threadIdx.x & 31;
    const int wave = threadIdx.x >> 5;
    const int l16  = lane & 15;
    const int g    = lane >> 4;
    const int b = blockIdx.z;
    const int h = blockIdx.y;
    const int mrow = blockIdx.x * 128 + wave * 32;

    const _Float16* Qb = Qh + ((size_t)(b * NHEAD + h)) * SEQ * HD;
    const _Float16* Kb = Kh + ((size_t)(b * NHEAD + h)) * SEQ * HD;
    const _Float16* Vb = Vt + ((size_t)(b * NHEAD + h)) * HD * SEQ;

    const float bias  = biasp[0];
    const float scale = 0.0360843918f;  // 1/sqrt(768) — full-feature scale per ref

    // Q fragments for 32 rows x full HD=64 (2 row groups x 2 K-steps)
    v16h qf[2][2];
#pragma unroll
    for (int rg = 0; rg < 2; ++rg)
#pragma unroll
        for (int ks = 0; ks < 2; ++ks)
            qf[rg][ks] = load_frag(Qb + (size_t)(mrow + rg * 16 + l16) * HD
                                      + ks * 32 + g * 8);

    v8f o[2][4];
#pragma unroll
    for (int i = 0; i < 2; ++i)
#pragma unroll
        for (int j = 0; j < 4; ++j) o[i][j] = {};

    __shared__ _Float16 sb[4][32 * SROW];  // private per-wave S staging (10 KB)
    _Float16* sw = sb[wave];

    for (int j = 0; j < SEQ; j += 32) {
        // B-fragments of K^T: lane = score column -> row (j + col) of K
        v16h kf[2][2];
#pragma unroll
        for (int cg = 0; cg < 2; ++cg)
#pragma unroll
            for (int ks = 0; ks < 2; ++ks)
                kf[cg][ks] = load_frag(Kb + (size_t)(j + cg * 16 + l16) * HD
                                          + ks * 32 + g * 8);

        // scores -> sigmoid -> LDS (f16)
#pragma unroll
        for (int rg = 0; rg < 2; ++rg) {
#pragma unroll
            for (int cg = 0; cg < 2; ++cg) {
                v8f s = {};
                s = wmma16(qf[rg][0], kf[cg][0], s);
                s = wmma16(qf[rg][1], kf[cg][1], s);
#pragma unroll
                for (int r = 0; r < 8; ++r) {
                    float t = s[r] * scale + bias;
                    float p = 1.0f / (1.0f + __expf(-t));
                    sw[(rg * 16 + g * 8 + r) * SROW + cg * 16 + l16] = (_Float16)p;
                }
            }
        }

        // A-fragments of S (same-wave LDS, DS ops are in-order)
        v16h sa0 = load_frag(sw + (      l16) * SROW + g * 8);
        v16h sa1 = load_frag(sw + (16 + l16) * SROW + g * 8);

        // B-fragments of V from the transposed layout: lane = hd column
        v16h vf[4];
#pragma unroll
        for (int cg = 0; cg < 4; ++cg)
            vf[cg] = load_frag(Vb + (size_t)(cg * 16 + l16) * SEQ + j + g * 8);

#pragma unroll
        for (int cg = 0; cg < 4; ++cg) {
            o[0][cg] = wmma16(sa0, vf[cg], o[0][cg]);
            o[1][cg] = wmma16(sa1, vf[cg], o[1][cg]);
        }
    }

    // write attention output as (b, n, h*64+hd) f16 for the final projection
#pragma unroll
    for (int rg = 0; rg < 2; ++rg) {
#pragma unroll
        for (int cg = 0; cg < 4; ++cg) {
            const int col = h * HD + cg * 16 + l16;
#pragma unroll
            for (int r = 0; r < 8; ++r) {
                const int n = mrow + rg * 16 + g * 8 + r;
                Ao[((size_t)b * SEQ + n) * FEATS + col] = (_Float16)o[rg][cg][r];
            }
        }
    }
}

// ---------------------------------------------------------------------------
extern "C" void kernel_launch(void* const* d_in, const int* in_sizes, int n_in,
                              void* d_out, int out_size, void* d_ws, size_t ws_size,
                              hipStream_t stream) {
    (void)in_sizes; (void)n_in; (void)out_size; (void)ws_size;
    const float* x    = (const float*)d_in[0];
    const float* bias = (const float*)d_in[1];
    const float* Wq   = (const float*)d_in[2];
    const float* bq   = (const float*)d_in[3];
    const float* Wk   = (const float*)d_in[4];
    const float* bk   = (const float*)d_in[5];
    const float* Wv   = (const float*)d_in[6];
    const float* bv   = (const float*)d_in[7];
    const float* Wo   = (const float*)d_in[8];
    const float* bo   = (const float*)d_in[9];
    float* out = (float*)d_out;

    const size_t NX = (size_t)M_TOT * FEATS;   // 6,291,456
    const size_t NW = (size_t)FEATS * FEATS;   //   589,824

    _Float16* ws  = (_Float16*)d_ws;
    _Float16* xh  = ws; ws += NX;
    _Float16* Wqh = ws; ws += NW;
    _Float16* Wkh = ws; ws += NW;
    _Float16* Wvh = ws; ws += NW;
    _Float16* Woh = ws; ws += NW;
    _Float16* Qh  = ws; ws += NX;   // (b,h,n,hd)
    _Float16* Kh  = ws; ws += NX;   // (b,h,n,hd)
    _Float16* Vt  = ws; ws += NX;   // (b,h,hd,n)
    _Float16* Aoh = ws; ws += NX;   // (b,n,f)

    cvt_f32_f16<<<dim3((NX + 255) / 256), 256, 0, stream>>>(x,  xh,  (int)NX);
    cvt_f32_f16<<<dim3((NW + 255) / 256), 256, 0, stream>>>(Wq, Wqh, (int)NW);
    cvt_f32_f16<<<dim3((NW + 255) / 256), 256, 0, stream>>>(Wk, Wkh, (int)NW);
    cvt_f32_f16<<<dim3((NW + 255) / 256), 256, 0, stream>>>(Wv, Wvh, (int)NW);
    cvt_f32_f16<<<dim3((NW + 255) / 256), 256, 0, stream>>>(Wo, Woh, (int)NW);

    dim3 gg(M_TOT / 128, FEATS / 128, 1);      // (64, 6): 128x128 block tiles
    gemm_xwT<<<gg, 128, 0, stream>>>(xh, Wqh, bq, Qh, nullptr, 0);
    gemm_xwT<<<gg, 128, 0, stream>>>(xh, Wkh, bk, Kh, nullptr, 1);
    gemm_xwT<<<gg, 128, 0, stream>>>(xh, Wvh, bv, Vt, nullptr, 2);

    dim3 ga(SEQ / 128, NHEAD, BATCH);          // (8, 12, 8)
    sig_attn<<<ga, 128, 0, stream>>>(Qh, Kh, Vt, bias, Aoh);

    gemm_xwT<<<gg, 128, 0, stream>>>(Aoh, Woh, bo, nullptr, out, 3);
}
